// LSTMSingleStep_14053132992744
// MI455X (gfx1250) — compile-verified, matrix operand order checked by reading, not run on hardware
//
#include <hip/hip_runtime.h>
#include <hip/hip_bf16.h>
#include <math.h>

typedef __attribute__((ext_vector_type(16))) __bf16 v16bf;
typedef __attribute__((ext_vector_type(8)))  float  v8f;

#define Bsz 1024
#define Ssz 128
#define SDz 8
#define Dz  128
#define Lz  6

union U8 { v8f v; float f[8]; };

__device__ __forceinline__ float sigm(float x) { return 1.0f / (1.0f + __expf(-x)); }
__device__ __forceinline__ unsigned short f2bf(float f) {
    unsigned int u = __float_as_uint(f);
    unsigned int r = (u + 0x7FFFu + ((u >> 16) & 1u)) >> 16;
    return (unsigned short)r;
}

// ---------------------------------------------------------------------------
// Kernel 1: pack cell_w [L][256][512] f32 -> bf16 tiles in WMMA-B layout.
// Tile (l, kt, nt) is 32(K)x16(N): lane p (0..31) holds 16 contiguous bf16:
//   lanes 0-15 : N = nt*16 + lane,      K = kt*32 + j      (j = 0..15)
//   lanes 16-31: N = nt*16 + (lane-16), K = kt*32 + 16 + j
// ---------------------------------------------------------------------------
__global__ __launch_bounds__(256) void pack_w_kernel(const float* __restrict__ cw,
                                                     unsigned short* __restrict__ wpack) {
    int idx = blockIdx.x * 256 + threadIdx.x;            // 0 .. L*8*32*512-1
    int j    =  idx        & 15;
    int lane = (idx >> 4)  & 31;
    int nt   = (idx >> 9)  & 31;
    int kt   = (idx >> 14) & 7;
    int l    =  idx >> 17;
    int k = kt * 32 + (lane & 16) + j;
    int n = nt * 16 + (lane & 15);
    float v = cw[(size_t)l * 256 * 512 + (size_t)k * 512 + n];
    wpack[idx] = f2bf(v);
}

// ---------------------------------------------------------------------------
// Kernel 2: embedding  x = (GELU(LN(states@W1+b1)))@W2+b2.
// Output written DIRECTLY in WMMA-A tile-stream layout:
//   xpk[(((bt*S + t)*4 + kt)*512) + lane*16 + j]
// where bt = b/16, r = b%16, kt = d/32, lane = r + (d & 16), j = d & 15.
// ---------------------------------------------------------------------------
__device__ __forceinline__ float blockReduce128(float v, float* red, int tid) {
    red[tid] = v; __syncthreads();
    for (int s = 64; s > 0; s >>= 1) {
        if (tid < s) red[tid] += red[tid + s];
        __syncthreads();
    }
    float r = red[0]; __syncthreads();
    return r;
}

__global__ __launch_bounds__(128) void embed_kernel(const float* __restrict__ states,
                                                    const float* __restrict__ w1,
                                                    const float* __restrict__ b1,
                                                    const float* __restrict__ lnw,
                                                    const float* __restrict__ lnb,
                                                    const float* __restrict__ w2,
                                                    const float* __restrict__ b2,
                                                    unsigned short* __restrict__ xpk) {
    __shared__ float w2s[Dz * Dz];
    __shared__ float svec[SDz];
    __shared__ float gv[Dz];
    __shared__ float red[Dz];
    int tid = threadIdx.x;
    for (int i = tid; i < Dz * Dz; i += 128) w2s[i] = w2[i];
    __syncthreads();

    int kt = tid >> 5;
    int j  = tid & 15;

    for (int rr = 0; rr < 16; ++rr) {
        size_t row = (size_t)blockIdx.x * 16 + rr;       // flattened (b, s)
        int b = (int)(row >> 7);                          // row / S
        int s = (int)(row & 127);                         // row % S
        if (tid < SDz) svec[tid] = states[row * SDz + tid];
        __syncthreads();
        float t1 = b1[tid];
        #pragma unroll
        for (int k = 0; k < SDz; ++k) t1 += svec[k] * w1[k * Dz + tid];
        float m  = blockReduce128(t1, red, tid) * (1.0f / Dz);
        float dv = t1 - m;
        float var = blockReduce128(dv * dv, red, tid) * (1.0f / Dz);
        float x = dv * rsqrtf(var + 1e-5f) * lnw[tid] + lnb[tid];
        float g = 0.5f * x * (1.0f + erff(x * 0.70710678118f));
        gv[tid] = g;
        __syncthreads();
        float t2 = b2[tid];
        for (int k = 0; k < Dz; ++k) t2 += gv[k] * w2s[k * Dz + tid];
        int bt = b >> 4, r = b & 15;
        int lane_p = r + (tid & 16);
        size_t off = ((((size_t)bt * Ssz + s) * 4 + kt) * 512) + lane_p * 16 + j;
        xpk[off] = f2bf(t2);
        __syncthreads();
    }
}

// ---------------------------------------------------------------------------
// Kernel 3: one LSTM layer scanned over S=128 timesteps. One block = 16 batch
// rows, 8 wave32s. Layer weights (32 B-tiles = 256 VGPRs/wave) loaded once,
// register-resident across the whole scan. Global inp A-tiles are software-
// pipelined one timestep ahead (+L2 prefetch two ahead); LDS h-tiles loaded as
// one clause. Wave w owns gate columns [16w,16w+16) for all 4 gates.
// ---------------------------------------------------------------------------
__global__ __launch_bounds__(256, 1) void layer_kernel(const unsigned short* __restrict__ inpk,
                                                       const unsigned short* __restrict__ wl_,
                                                       const float* __restrict__ bias,
                                                       unsigned short* __restrict__ outpk,
                                                       int is_last,
                                                       const float* __restrict__ olnw,
                                                       const float* __restrict__ olnb,
                                                       const float* __restrict__ ow1,
                                                       const float* __restrict__ ob1,
                                                       const float* __restrict__ ow2,
                                                       const float* __restrict__ ob2,
                                                       float* __restrict__ out) {
    __shared__ __align__(16) __bf16 hpk[4 * 512];        // h_t in A-tile layout (4 KB)
    __shared__ float cst[16 * Dz];                       // c state (8 KB)
    __shared__ float hfin[16 * Dz];                      // head input (8 KB)
    __shared__ float headA[16 * Dz];                     // head scratch (8 KB)

    const __bf16* wl = (const __bf16*)wl_;
    int tid  = threadIdx.x;
    int lane = tid & 31;
    int w    = tid >> 5;
    int bid  = blockIdx.x;
    int cl   = lane & 15;
    int hi   = lane >> 4;

    for (int i = tid; i < 4 * 512; i += 256) hpk[i] = (__bf16)0.0f;
    for (int i = tid; i < 16 * Dz; i += 256) cst[i] = 0.0f;
    __syncthreads();

    // ---- hoist this wave's 32 weight tiles into registers (256 VGPRs) ----
    v16bf wreg[8][4];
    #pragma unroll
    for (int kt = 0; kt < 8; ++kt)
        #pragma unroll
        for (int g = 0; g < 4; ++g) {
            int nt = w + g * 8;
            wreg[kt][g] = *(const v16bf*)(wl + ((size_t)kt * 32 + nt) * 512 + lane * 16);
        }

    const int dcol = w * 16 + cl;
    const int kth  = dcol >> 5;
    const int lpw  = dcol & 16;
    const int jw   = dcol & 15;
    const float bI = bias[dcol];
    const float bF = bias[128 + dcol];
    const float bG = bias[256 + dcol];
    const float bO = bias[384 + dcol];

    const __bf16* inb = (const __bf16*)inpk + ((size_t)bid * Ssz * 4) * 512;

    // ---- prime the A-tile pipeline with t = 0 ----
    v16bf a_cur[4];
    #pragma unroll
    for (int kt = 0; kt < 4; ++kt)
        a_cur[kt] = *(const v16bf*)(inb + kt * 512 + lane * 16);

    for (int t = 0; t < Ssz; ++t) {
        // ---- issue next timestep's inp tiles now; consumed next iteration ----
        int tn = (t < Ssz - 1) ? t + 1 : t;
        const __bf16* nb = inb + (size_t)tn * 4 * 512;
        v16bf a_nxt[4];
        #pragma unroll
        for (int kt = 0; kt < 4; ++kt)
            a_nxt[kt] = *(const v16bf*)(nb + kt * 512 + lane * 16);
        // ---- L2 prefetch two timesteps ahead (one 128B line per lane) ----
        int tp = (t < Ssz - 2) ? t + 2 : t;
        __builtin_prefetch((const void*)(inb + (size_t)tp * 4 * 512 + lane * 64), 0, 1);

        // ---- load all four h A-tiles as one clause ----
        v16bf ah[4];
        #pragma unroll
        for (int kt = 0; kt < 4; ++kt)
            ah[kt] = *(const v16bf*)((const __bf16*)hpk + kt * 512 + lane * 16);

        U8 acc[4];
        #pragma unroll
        for (int g = 0; g < 4; ++g)
            #pragma unroll
            for (int r = 0; r < 8; ++r) acc[g].f[r] = 0.0f;

        // ---- 32 back-to-back WMMAs: K 0..127 from a_cur, K 128..255 from ah ----
        #pragma unroll
        for (int kt = 0; kt < 4; ++kt)
            #pragma unroll
            for (int g = 0; g < 4; ++g)
                acc[g].v = __builtin_amdgcn_wmma_f32_16x16x32_bf16(
                    false, a_cur[kt], false, wreg[kt][g], (short)0, acc[g].v, false, false);
        #pragma unroll
        for (int kt = 0; kt < 4; ++kt)
            #pragma unroll
            for (int g = 0; g < 4; ++g)
                acc[g].v = __builtin_amdgcn_wmma_f32_16x16x32_bf16(
                    false, ah[kt], false, wreg[kt + 4][g], (short)0, acc[g].v, false, false);
        __syncthreads();   // all reads of hpk done before overwrite

        #pragma unroll
        for (int r = 0; r < 8; ++r) {
            int row = r + hi * 8;                         // C/D layout: lanes 16-31 -> M+8
            float gi = acc[0].f[r] + bI;
            float gf = acc[1].f[r] + bF;
            float gg = acc[2].f[r] + bG;
            float go = acc[3].f[r] + bO;
            float cold = cst[row * Dz + dcol];
            float cn = sigm(gf) * cold + sigm(gi) * tanhf(gg);
            float hn = sigm(go) * tanhf(cn);
            cst[row * Dz + dcol] = cn;
            hpk[kth * 512 + (row + lpw) * 16 + jw] = (__bf16)hn;
            if (is_last && t == Ssz - 1) hfin[row * Dz + dcol] = hn;
        }
        __syncthreads();   // writes visible to all waves

        // ---- stream h_t (A layout) to global for the next layer ----
        if (!is_last) {
            uint4 v = *(const uint4*)((const unsigned short*)hpk + tid * 8);
            *(uint4*)(outpk + ((size_t)(bid * Ssz + t) * 4) * 512 + tid * 8) = v;
        }

        #pragma unroll
        for (int kt = 0; kt < 4; ++kt) a_cur[kt] = a_nxt[kt];
    }

    // ---- output head: LN -> Linear(128) -> GELU -> Linear(8) on h @ t=S-1 ----
    if (is_last && tid < 16) {
        int row = tid;
        float m = 0.0f;
        for (int k = 0; k < Dz; ++k) m += hfin[row * Dz + k];
        m *= (1.0f / Dz);
        float v = 0.0f;
        for (int k = 0; k < Dz; ++k) { float d = hfin[row * Dz + k] - m; v += d * d; }
        float inv = rsqrtf(v * (1.0f / Dz) + 1e-5f);
        for (int k = 0; k < Dz; ++k)
            headA[row * Dz + k] = (hfin[row * Dz + k] - m) * inv * olnw[k] + olnb[k];
        for (int d = 0; d < Dz; ++d) {
            float a = ob1[d];
            for (int k = 0; k < Dz; ++k) a += headA[row * Dz + k] * ow1[k * Dz + d];
            hfin[row * Dz + d] = 0.5f * a * (1.0f + erff(a * 0.70710678118f));
        }
        for (int j = 0; j < SDz; ++j) {
            float a = ob2[j];
            for (int k = 0; k < Dz; ++k) a += hfin[row * Dz + k] * ow2[k * SDz + j];
            out[((size_t)bid * 16 + row) * SDz + j] = a;
        }
    }
}

// ---------------------------------------------------------------------------
extern "C" void kernel_launch(void* const* d_in, const int* in_sizes, int n_in,
                              void* d_out, int out_size, void* d_ws, size_t ws_size,
                              hipStream_t stream) {
    const float* states  = (const float*)d_in[0];
    const float* emb_w1  = (const float*)d_in[1];
    const float* emb_b1  = (const float*)d_in[2];
    const float* emb_lnw = (const float*)d_in[3];
    const float* emb_lnb = (const float*)d_in[4];
    const float* emb_w2  = (const float*)d_in[5];
    const float* emb_b2  = (const float*)d_in[6];
    const float* cell_w  = (const float*)d_in[7];
    const float* cell_b  = (const float*)d_in[8];
    const float* out_lnw = (const float*)d_in[9];
    const float* out_lnb = (const float*)d_in[10];
    const float* out_w1  = (const float*)d_in[11];
    const float* out_b1  = (const float*)d_in[12];
    const float* out_w2  = (const float*)d_in[13];
    const float* out_b2  = (const float*)d_in[14];
    float* out = (float*)d_out;

    char* ws = (char*)d_ws;
    unsigned short* wpack = (unsigned short*)ws;                        // 1.5 MB
    unsigned short* xpk   = (unsigned short*)(ws + (size_t)(2u  << 20)); // 32 MB
    unsigned short* buf0  = (unsigned short*)(ws + (size_t)(36u << 20)); // 32 MB

    // 1) pack cell weights to bf16 WMMA-B tiles (L*8*32*512 = 786432 elems)
    pack_w_kernel<<<786432 / 256, 256, 0, stream>>>(cell_w, wpack);

    // 2) embedding MLP -> x bf16, emitted directly in A-tile stream layout
    embed_kernel<<<(Bsz * Ssz) / 16, 128, 0, stream>>>(
        states, emb_w1, emb_b1, emb_lnw, emb_lnb, emb_w2, emb_b2, xpk);

    // 3) six layer passes, weights register-resident per pass; ping-pong xpk/buf0
    for (int l = 0; l < Lz; ++l) {
        const unsigned short* in  = (l & 1) ? buf0 : xpk;
        unsigned short*       o   = (l & 1) ? xpk  : buf0;
        int is_last = (l == Lz - 1);
        layer_kernel<<<Bsz / 16, 256, 0, stream>>>(
            in, wpack + (size_t)l * 8 * 32 * 512, cell_b + (size_t)l * 512, o, is_last,
            out_lnw, out_lnb, out_w1, out_b1, out_w2, out_b2, out);
    }
}
